// JointGraphAttention_75282186764970
// MI455X (gfx1250) — matrix operand order repaired and checked by reading, not compiled, not run on hardware
//
#include <hip/hip_runtime.h>
#include <hip/hip_bf16.h>
#include <math.h>

#define BB 2
#define NN 256
#define MM 512
#define CC 256
#define HH 8
#define DH 32
#define FF 256

typedef _Float16 v16h __attribute__((ext_vector_type(16)));
typedef _Float16 v8h  __attribute__((ext_vector_type(8)));
typedef float    v8f  __attribute__((ext_vector_type(8)));

// ---------------------------------------------------------------------------
// Kernel 0: repack W1/W2 into f16 WMMA-B-layout tiles; transpose Wq/Wk/Wv/Wo.
// B-layout (16-bit, 32x16, wave32): lane l -> column n = l&15; K = 8*(l>>4)
// + 2*vgpr + halfsel + 8*(vgpr>=4).  Fragment (kt,nt) stored as 32 lanes x 16 halves so a
// lane loads its fragment with a single contiguous 32-byte v16h load.
// ---------------------------------------------------------------------------
__global__ void pack_weights(const float* __restrict__ W1, const float* __restrict__ W2,
                             const float* __restrict__ Wq, const float* __restrict__ Wk,
                             const float* __restrict__ Wv, const float* __restrict__ Wo,
                             _Float16* __restrict__ W1p, _Float16* __restrict__ W2p,
                             float* __restrict__ WqT, float* __restrict__ WkT,
                             float* __restrict__ WvT, float* __restrict__ WoT) {
    const int tid  = blockIdx.x * blockDim.x + threadIdx.x;   // 0..65535
    const int task = blockIdx.y;
    if (task < 2) {
        const float* W  = (task == 0) ? W1 : W2;
        _Float16*    Wp = (task == 0) ? W1p : W2p;
        const int fi   = tid >> 9;        // fragment index = kt*16+nt
        const int rem  = tid & 511;
        const int lane = rem >> 4;
        const int e    = rem & 15;        // half index within lane's v16h
        const int kt   = fi >> 4;
        const int nt   = fi & 15;
        const int ncol = lane & 15;
        const int hi   = lane >> 4;
        const int vv   = e >> 1;
        const int hp   = e & 1;
        const int k    = (vv < 4 ? 2 * vv : 16 + 2 * (vv - 4)) + 8 * hi + hp;     // 0..31
        const int c    = nt * 16 + ncol;            // output channel
        const int f    = kt * 32 + k;               // input feature
        Wp[tid] = (_Float16)W[c * 256 + f];
    } else {
        const float* W  = (task == 2) ? Wq : (task == 3) ? Wk : (task == 4) ? Wv : Wo;
        float*       WT = (task == 2) ? WqT : (task == 3) ? WkT : (task == 4) ? WvT : WoT;
        const int c = tid & 255;
        const int j = tid >> 8;
        WT[j * 256 + c] = W[c * 256 + j];
    }
}

// ---------------------------------------------------------------------------
// Kernel 1: q/k/v projections (small; VALU matvec with transposed weights).
// ---------------------------------------------------------------------------
__global__ void qkv_proj(const float* __restrict__ query, const float* __restrict__ key,
                         const float* __restrict__ WqT, const float* __restrict__ WkT,
                         const float* __restrict__ WvT,
                         const float* __restrict__ bq, const float* __restrict__ bv,
                         float* __restrict__ qo, float* __restrict__ ko,
                         float* __restrict__ vo) {
    __shared__ float s_in[CC];
    const int blk = blockIdx.x;
    const float* in; const float* WT; const float* bias; float* out;
    if (blk < BB * NN)                { in = query + blk * CC;                 WT = WqT; bias = bq;      out = qo + blk * CC; }
    else if (blk < BB * NN + BB * MM) { int r = blk - BB * NN;  in = key + r * CC; WT = WkT; bias = nullptr; out = ko + r * CC; }
    else                              { int r = blk - BB * NN - BB * MM; in = key + r * CC; WT = WvT; bias = bv; out = vo + r * CC; }
    const int c = threadIdx.x;
    s_in[c] = in[c];
    __syncthreads();
    float acc = bias ? bias[c] : 0.f;
#pragma unroll 8
    for (int j = 0; j < CC; ++j) acc += s_in[j] * WT[j * CC + c];
    out[c] = acc;
}

// ---------------------------------------------------------------------------
// Kernel 2: fused positional-MLP (WMMA f16) + gated scores + softmax + AV +
// output projection + residual.  One workgroup per (b,n); 8 waves x 16 m-rows
// per pass; 4 passes cover M=512.
//
// Weight fragments are addressed through an asm-laundered integer offset
// (keeps global address-space provenance -> global_load_b128, but blocks
// LICM from hoisting/spilling the loop-invariant weight stream), and the
// K-loop is explicitly double-buffered so each WMMA overlaps the next
// fragment's L2 load instead of draining LOADcnt to zero.
// ---------------------------------------------------------------------------
__global__ void __launch_bounds__(256)
jga_main(const float* __restrict__ query, const float* __restrict__ query_pos,
         const float* __restrict__ freqs,
         const float* __restrict__ b1, const float* __restrict__ b2,
         const float* __restrict__ bo,
         const _Float16* __restrict__ W1p, const _Float16* __restrict__ W2p,
         const float* __restrict__ qbuf, const float* __restrict__ kbuf,
         const float* __restrict__ vbuf, const float* __restrict__ WoT,
         float* __restrict__ out) {
    __shared__ float s_freqs[FF / 2];
    __shared__ float s_q[CC];
    __shared__ float s_b1[CC];
    __shared__ float s_b2[CC];
    __shared__ __align__(16) _Float16 hm2[8][16][32];   // per-wave hmid chunk
    __shared__ float s_scores[MM * HH];                 // 16 KB
    __shared__ float s_x[CC];

    const int b = blockIdx.x / NN;
    const int n = blockIdx.x % NN;
    const int tid    = threadIdx.x;
    const int w      = tid >> 5;
    const int lane   = tid & 31;
    const int lanelo = lane & 15;
    const int hi     = lane >> 4;

    if (tid < 128) s_freqs[tid] = freqs[tid];
    s_q[tid]  = qbuf[(b * NN + n) * CC + tid];
    s_b1[tid] = b1[tid];
    s_b2[tid] = b2[tid];
    __syncthreads();

    const float scale = 0.17677669529663687f;   // 32^-0.5
    const v8f zero8 = {0.f, 0.f, 0.f, 0.f, 0.f, 0.f, 0.f, 0.f};

#pragma unroll 1
    for (int pass = 0; pass < 4; ++pass) {
        const int m_base = pass * 128;
        const int mrow   = m_base + w * 16 + lanelo;       // A-layout row
        const float t    = query_pos[(b * NN + n) * MM + mrow];

        // ---- build emb A-fragments in ISA A-register layout (no emb buffer)
        v16h a[8];
#pragma unroll
        for (int kt = 0; kt < 8; ++kt) {
#pragma unroll
            for (int e = 0; e < 16; ++e) {
                const int vv = e >> 1, hp = e & 1;
                const int k  = (vv < 4 ? 2 * vv : 16 + 2 * (vv - 4)) + 8 * hi + hp;
                const int f  = kt * 32 + k;
                float val;
                if (kt < 4) val = __cosf(t * s_freqs[f]);
                else        val = __sinf(t * s_freqs[f - 128]);
                a[kt][e] = (_Float16)val;
            }
        }

        // ---- phase 1: layer-1 GEMM, build layer-2 A-frags through LDS
        v16h a2[8];
#pragma unroll
        for (int pt = 0; pt < 8; ++pt) {           // 32-channel chunk of hmid
#pragma unroll
            for (int sub = 0; sub < 2; ++sub) {
                const int nt = pt * 2 + sub;
                unsigned o1 = 0;
                asm volatile("" : "+s"(o1));       // anti-hoist fence (opaque 0)
                const v16h* wb = (const v16h*)(W1p + o1) + (nt * 32 + lane);
                v8f acc1 = zero8;
                v16h cur = wb[0];                  // frag(kt=0)
#pragma unroll
                for (int kt = 0; kt < 8; ++kt) {   // frag(kt) = wb[kt*512]
                    v16h nxt = cur;
                    if (kt < 7) nxt = wb[(kt + 1) * 512];
                    acc1 = __builtin_amdgcn_wmma_f32_16x16x32_f16(
                        false, a[kt], false, cur, (short)0, acc1, false, false);
                    cur = nxt;
                }
                const int c = nt * 16 + lanelo;
                const float bb1 = s_b1[c];
#pragma unroll
                for (int r = 0; r < 8; ++r) {      // SiLU + stage to LDS
                    float x  = acc1[r] + bb1;
                    float sx = x * __builtin_amdgcn_rcpf(1.f + __expf(-x));
                    hm2[w][r + 8 * hi][sub * 16 + lanelo] = (_Float16)sx;
                }
            }
            // re-read own chunk in A-layout (two 16-B LDS loads)
            const _Float16* rowp = &hm2[w][lanelo][0];
            v8h alo = *(const v8h*)(rowp + 8 * hi);
            v8h ahi = *(const v8h*)(rowp + 16 + 8 * hi);
#pragma unroll
            for (int e = 0; e < 8; ++e) { a2[pt][e] = alo[e]; a2[pt][e + 8] = ahi[e]; }
        }

        // ---- phase 2: layer-2 GEMM + gated scores, two N-halves
        const float* krow0 = kbuf + (b * MM + m_base + w * 16) * CC;
#pragma unroll
        for (int half = 0; half < 2; ++half) {
            v8f acc[8];
#pragma unroll
            for (int j = 0; j < 8; ++j) acc[j] = zero8;
#pragma unroll
            for (int j = 0; j < 8; ++j) {
                const int nt2 = half * 8 + j;
                unsigned o2 = 0;
                asm volatile("" : "+s"(o2));       // anti-hoist fence (opaque 0)
                const v16h* wb = (const v16h*)(W2p + o2) + (nt2 * 32 + lane);
                v16h cur = wb[0];                  // frag(pt=0)
#pragma unroll
                for (int pt = 0; pt < 8; ++pt) {   // frag(pt) = wb[pt*512]
                    v16h nxt = cur;
                    if (pt < 7) nxt = wb[(pt + 1) * 512];
                    acc[j] = __builtin_amdgcn_wmma_f32_16x16x32_f16(
                        false, a2[pt], false, cur, (short)0, acc[j], false, false);
                    cur = nxt;
                }
            }
            // per-lane channel constants for this half's 8 tiles
            float qc[8], bc[8];
#pragma unroll
            for (int j = 0; j < 8; ++j) {
                const int c = (half * 8 + j) * 16 + lanelo;
                qc[j] = s_q[c];
                bc[j] = s_b2[c];
            }
            // scores for heads [half*4, half*4+4)
#pragma unroll
            for (int r = 0; r < 8; ++r) {
                const int mr = r + 8 * hi;                       // D-layout row
                const int m  = m_base + w * 16 + mr;
                const float* krow = krow0 + mr * CC + half * 128 + lanelo;
#pragma unroll
                for (int hh = 0; hh < 4; ++hh) {
                    const int j0 = 2 * hh, j1 = 2 * hh + 1;
                    float v0 = (acc[j0][r] + bc[j0]) * qc[j0] * krow[j0 * 16];
                    float v1 = (acc[j1][r] + bc[j1]) * qc[j1] * krow[j1 * 16];
                    float val = v0 + v1;
#pragma unroll
                    for (int off = 1; off < 16; off <<= 1)
                        val += __shfl_xor(val, off, 16);         // half-wave butterfly
                    const int h = half * 4 + hh;
                    if (lanelo == h) s_scores[m * HH + h] = val * scale;
                }
            }
        }
    }
    __syncthreads();

    // ---- softmax over m per head (one wave per head)
    {
        const int h = w;
        float mx = -1e30f;
        for (int m = lane; m < MM; m += 32) mx = fmaxf(mx, s_scores[m * HH + h]);
#pragma unroll
        for (int off = 1; off < 32; off <<= 1) mx = fmaxf(mx, __shfl_xor(mx, off, 32));
        float sum = 0.f;
        for (int m = lane; m < MM; m += 32) {
            float e = __expf(s_scores[m * HH + h] - mx);
            s_scores[m * HH + h] = e;
            sum += e;
        }
#pragma unroll
        for (int off = 1; off < 32; off <<= 1) sum += __shfl_xor(sum, off, 32);
        const float inv = __builtin_amdgcn_rcpf(sum);
        for (int m = lane; m < MM; m += 32) s_scores[m * HH + h] *= inv;
    }
    __syncthreads();

    // ---- x = attn @ v   (coalesced v streaming from L2)
    {
        const int c = tid;
        const int h = c >> 5;
        float acc = 0.f;
        for (int m = 0; m < MM; ++m) acc += s_scores[m * HH + h] * vbuf[(b * MM + m) * CC + c];
        s_x[c] = acc;
    }
    __syncthreads();

    // ---- out = x @ Wo^T + bo + residual
    {
        const int c = tid;
        float acc = bo[c] + query[(b * NN + n) * CC + c];
#pragma unroll 8
        for (int j = 0; j < CC; ++j) acc += s_x[j] * WoT[j * CC + c];
        out[(b * NN + n) * CC + c] = acc;
    }
}

// ---------------------------------------------------------------------------
extern "C" void kernel_launch(void* const* d_in, const int* in_sizes, int n_in,
                              void* d_out, int out_size, void* d_ws, size_t ws_size,
                              hipStream_t stream) {
    (void)in_sizes; (void)n_in; (void)out_size; (void)ws_size;
    const float* query = (const float*)d_in[0];
    const float* key   = (const float*)d_in[1];
    const float* qpos  = (const float*)d_in[2];
    const float* Wq    = (const float*)d_in[3];
    const float* bq    = (const float*)d_in[4];
    const float* Wk    = (const float*)d_in[5];
    const float* Wv    = (const float*)d_in[6];
    const float* bv    = (const float*)d_in[7];
    const float* Wo    = (const float*)d_in[8];
    const float* bo    = (const float*)d_in[9];
    const float* W1    = (const float*)d_in[10];
    const float* b1    = (const float*)d_in[11];
    const float* W2    = (const float*)d_in[12];
    const float* b2    = (const float*)d_in[13];
    const float* freqs = (const float*)d_in[14];

    char* ws = (char*)d_ws;
    _Float16* W1p = (_Float16*)(ws + 0);          // 128 KB
    _Float16* W2p = (_Float16*)(ws + 131072);     // 128 KB
    float* WqT = (float*)(ws + 262144);           // 256 KB each
    float* WkT = (float*)(ws + 524288);
    float* WvT = (float*)(ws + 786432);
    float* WoT = (float*)(ws + 1048576);
    float* qb  = (float*)(ws + 1310720);          // B*N*C f32
    float* kb  = (float*)(ws + 1835008);          // B*M*C f32
    float* vb  = (float*)(ws + 2883584);          // B*M*C f32

    pack_weights<<<dim3(256, 6), 256, 0, stream>>>(W1, W2, Wq, Wk, Wv, Wo,
                                                   W1p, W2p, WqT, WkT, WvT, WoT);
    qkv_proj<<<BB * NN + 2 * BB * MM, 256, 0, stream>>>(query, key, WqT, WkT, WvT,
                                                        bq, bv, qb, kb, vb);
    jga_main<<<BB * NN, 256, 0, stream>>>(query, qpos, freqs, b1, b2, bo,
                                          W1p, W2p, qb, kb, vb, WoT, (float*)d_out);
}
